// GraphMambaGMN_56229711839358
// MI455X (gfx1250) — compile-verified, matrix operand order checked by reading, not run on hardware
//
#include <hip/hip_runtime.h>
#include <hip/hip_bf16.h>
#include <cstdint>
#include <cstddef>

// ---------------------------------------------------------------------------
// GraphMambaGMN for MI455X (gfx1250, wave32, WMMA).
// All einsums -> v_wmma_f32_16x16x32_f16 tile kernel:
//   - weight tile staged to LDS via GLOBAL_LOAD_ASYNC_TO_LDS_B128 (ASYNCcnt
//     path) when the toolchain exposes the async builtins, else direct loads
//   - b128 global loads for A, ds_load_b128 for B fragments
//   - bounds handled by clamp+mask (no divergent branches in the K loop)
// SSM scans stay f32 VALU with register-resident state.
// ---------------------------------------------------------------------------

#if __has_builtin(__builtin_amdgcn_global_load_async_to_lds_b128) && \
    __has_builtin(__builtin_amdgcn_s_wait_asynccnt)
#define GMN_ASYNC 1
#else
#define GMN_ASYNC 0
#endif

#define GMN_AS1 __attribute__((address_space(1)))
#define GMN_AS3 __attribute__((address_space(3)))

namespace cfg {
constexpr int B  = 4,  N  = 1024, H = 64, D = 64;
constexpr int L  = 8,  SUB = 16;
constexpr int DI = 128, DS = 16, DC = 4, R = 4;
constexpr int E  = 32768;
constexpr int M0 = B * N * L;   // 32768 level-0 tokens
constexpr int M1 = B * N;       // 4096 level-1 tokens
}

typedef __attribute__((ext_vector_type(16))) _Float16 v16h;
typedef __attribute__((ext_vector_type(8)))  _Float16 v8h;
typedef __attribute__((ext_vector_type(8)))  float    v8f;
typedef __attribute__((ext_vector_type(4)))  int      v4i;

// ---------------------------------------------------------------------------
// WMMA GEMM:  C[m,n] (+)= alpha * sum_k A[m,k] * W[n,k]  + bias[n]
// A: M x K f32 row-major (M % 64 == 0), W: N x K f32 row-major, K in {64,128}.
// grid = (ceil(N/16), M/64); block = 128 threads (4 waves).
// All 4 waves share blockIdx.x's 16-row weight tile staged to LDS as f16.
// ---------------------------------------------------------------------------
template<int ACT, int ACC>
__global__ void gmn_gemm(const float* __restrict__ A, const float* __restrict__ W,
                         const float* __restrict__ bias, float* __restrict__ C,
                         int M, int N, int K, float alpha)
{
    __shared__ _Float16 wtile[16 * 128];        // K <= 128

    const int tid   = threadIdx.x;
    const int lane  = tid & 31;
    const int wv    = tid >> 5;
    const int tn    = blockIdx.x;
    const int tm    = blockIdx.y * 4 + wv;
    const int nBase = tn << 4;

#if GMN_ASYNC
    // ---- async-stage the raw f32 tile into LDS (ASYNCcnt path) ----
    __shared__ float wtile_f32[16 * 128];
    for (int i = tid; i < 4 * K; i += 128) {     // one b128 chunk = 4 floats
        const int f  = i * 4;
        const int r  = f / K, k = f - r * K;
        const int n  = nBase + r;
        const int nc = (n < N) ? n : (N - 1);    // keep the address in-bounds
        __builtin_amdgcn_global_load_async_to_lds_b128(
            (GMN_AS1 v4i*)(W + (size_t)nc * K + k),
            (GMN_AS3 v4i*)(wtile_f32 + r * K + k),
            0, 0);
    }
    __builtin_amdgcn_s_wait_asynccnt(0);         // own wave's copies complete
    __syncthreads();                             // all waves' copies visible
    // masked f32 -> f16 conversion (zeroes out-of-range weight rows)
    for (int i = tid; i < 16 * K; i += 128) {
        const int   r   = i / K;
        const float msk = (nBase + r < N) ? 1.f : 0.f;
        wtile[i] = (_Float16)(wtile_f32[i] * msk);
    }
#else
    // ---- direct staging fallback: load, mask, convert ----
    for (int i = tid; i < 16 * K; i += 128) {
        const int r = i / K, k = i - r * K;
        const int n = nBase + r;
        const int nc = (n < N) ? n : (N - 1);
        const float msk = (n < N) ? 1.f : 0.f;
        wtile[r * K + k] = (_Float16)(W[(size_t)nc * K + k] * msk);
    }
#endif
    __syncthreads();

    const int hi = lane >> 4;                   // half-wave selector
    const int m  = (tm << 4) + (lane & 15);     // A row for this lane
    const int nl = lane & 15;                   // local weight row / C col

    const float*    arow = A + (size_t)m * K;
    const _Float16* wrow = wtile + nl * K;

    v8f acc = {};
    for (int k0 = 0; k0 < K; k0 += 32) {
        // A fragment: ISA K-pattern is contiguous in two groups of 8.
        const float4 a0 = *(const float4*)(arow + k0 + (hi << 3));
        const float4 a1 = *(const float4*)(arow + k0 + (hi << 3) + 4);
        const float4 a2 = *(const float4*)(arow + k0 + 16 + (hi << 3));
        const float4 a3 = *(const float4*)(arow + k0 + 16 + (hi << 3) + 4);
        v16h af;
        af[0]  = (_Float16)a0.x; af[1]  = (_Float16)a0.y;
        af[2]  = (_Float16)a0.z; af[3]  = (_Float16)a0.w;
        af[4]  = (_Float16)a1.x; af[5]  = (_Float16)a1.y;
        af[6]  = (_Float16)a1.z; af[7]  = (_Float16)a1.w;
        af[8]  = (_Float16)a2.x; af[9]  = (_Float16)a2.y;
        af[10] = (_Float16)a2.z; af[11] = (_Float16)a2.w;
        af[12] = (_Float16)a3.x; af[13] = (_Float16)a3.y;
        af[14] = (_Float16)a3.z; af[15] = (_Float16)a3.w;

        // B fragment: two ds_load_b128 from the staged tile.
        const v8h blo = *(const v8h*)(wrow + k0 + (hi << 3));
        const v8h bhi = *(const v8h*)(wrow + k0 + 16 + (hi << 3));
        const v16h bf = __builtin_shufflevector(blo, bhi,
                              0, 1, 2, 3, 4, 5, 6, 7, 8, 9, 10, 11, 12, 13, 14, 15);

        acc = __builtin_amdgcn_wmma_f32_16x16x32_f16(false, af, false, bf,
                                                     (short)0, acc, false, false);
    }

    const int n = nBase + nl;
    if (n < N) {
        const float bv = bias ? bias[n] : 0.f;
#pragma unroll
        for (int r = 0; r < 8; ++r) {
            const int   mr = (tm << 4) + r + (hi << 3);
            float v = alpha * acc[r] + bv;
            if (ACT == 1) v = fmaxf(v, 0.f);
            const size_t o = (size_t)mr * N + n;
            if (ACC) C[o] += v; else C[o] = v;
        }
    }
}

// ---------------------------------------------------------------------------
// pooled[b,n,l,h] = mean_s x[b, token_ids[b,n,l,s], h]
// ---------------------------------------------------------------------------
__global__ void gmn_pool(const float* __restrict__ x, const int* __restrict__ ids,
                         float* __restrict__ pooled)
{
    using namespace cfg;
    const int idx = blockIdx.x * blockDim.x + threadIdx.x;
    if (idx >= M0 * H) return;
    const int h   = idx & (H - 1);
    const int tok = idx / H;                 // (b*N+n)*L + l
    const int b   = tok / (N * L);
    const int* tp = ids + (size_t)tok * SUB;
    float s = 0.f;
#pragma unroll
    for (int u = 0; u < SUB; ++u)
        s += x[((size_t)b * N + tp[u]) * H + h];
    pooled[idx] = s * (1.f / SUB);
}

// ---------------------------------------------------------------------------
// Row LayerNorm over D=64 (in/out may alias; a thread owns a row).
// ---------------------------------------------------------------------------
__global__ void gmn_layernorm(const float* in, float* out,
                              const float* __restrict__ g, const float* __restrict__ bta,
                              int rows)
{
    using namespace cfg;
    const int row = blockIdx.x * blockDim.x + threadIdx.x;
    if (row >= rows) return;
    const float* p = in + (size_t)row * D;
    float s = 0.f, s2 = 0.f;
#pragma unroll 8
    for (int c = 0; c < D; ++c) { float v = p[c]; s += v; s2 += v * v; }
    const float m    = s * (1.f / D);
    const float var  = s2 * (1.f / D) - m * m;
    const float rstd = rsqrtf(var + 1e-5f);
    float* q = out + (size_t)row * D;
#pragma unroll 8
    for (int c = 0; c < D; ++c) q[c] = (p[c] - m) * rstd * g[c] + bta[c];
}

// LN with row gather (for permuted level-1 input).
__global__ void gmn_gather_ln(const float* __restrict__ node, const int* __restrict__ perm,
                              const float* __restrict__ g, const float* __restrict__ bta,
                              float* __restrict__ out)
{
    using namespace cfg;
    const int row = blockIdx.x * blockDim.x + threadIdx.x;
    if (row >= M1) return;
    const int b   = row / N;
    const float* p = node + ((size_t)b * N + perm[row]) * D;
    float s = 0.f, s2 = 0.f;
#pragma unroll 8
    for (int c = 0; c < D; ++c) { float v = p[c]; s += v; s2 += v * v; }
    const float m    = s * (1.f / D);
    const float rstd = rsqrtf(s2 * (1.f / D) - m * m + 1e-5f);
    float* q = out + (size_t)row * D;
#pragma unroll 8
    for (int c = 0; c < D; ++c) q[c] = (p[c] - m) * rstd * g[c] + bta[c];
}

// ---------------------------------------------------------------------------
// Depthwise causal conv (DC=4 taps) + SiLU. dir=0: taps from past (causal);
// dir=1: taps from future (realizes flip->conv->flip).
// ---------------------------------------------------------------------------
__global__ void gmn_conv_silu(const float* __restrict__ xz, const float* __restrict__ cw,
                              const float* __restrict__ cb, float* __restrict__ xc,
                              int Mtok, int T, int dir)
{
    using namespace cfg;
    const int idx = blockIdx.x * blockDim.x + threadIdx.x;
    if (idx >= Mtok * DI) return;
    const int d   = idx & (DI - 1);
    const int row = idx / DI;
    const int s   = row / T, t = row % T;
    float acc = cb[d];
#pragma unroll
    for (int k = 0; k < DC; ++k) {
        const int tt = dir ? (t + (DC - 1) - k) : (t - (DC - 1) + k);
        if (tt >= 0 && tt < T)
            acc += xz[((size_t)s * T + tt) * (2 * DI) + d] * cw[d * DC + k];
    }
    const float sig = 1.f / (1.f + __expf(-acc));
    xc[idx] = acc * sig;
}

// dt[t,d] = softplus( sum_r dbc[t,r] * Wdt[d,r] + bdt[d] )   (K=4: VALU)
__global__ void gmn_dt_softplus(const float* __restrict__ dbc, const float* __restrict__ Wdt,
                                const float* __restrict__ bdt, float* __restrict__ dt,
                                int Mtok)
{
    using namespace cfg;
    const int idx = blockIdx.x * blockDim.x + threadIdx.x;
    if (idx >= Mtok * DI) return;
    const int d   = idx & (DI - 1);
    const int row = idx / DI;
    const float* p = dbc + (size_t)row * (R + 2 * DS);
    float v = bdt[d];
#pragma unroll
    for (int r = 0; r < R; ++r) v += p[r] * Wdt[d * R + r];
    dt[idx] = (v > 20.f) ? v : log1pf(__expf(v));
}

// ---------------------------------------------------------------------------
// Selective-scan: thread per (sequence, channel d); 16-wide state in regs.
// Fuses +Dp*xc and the *silu(z) gate into the store.
// ---------------------------------------------------------------------------
__global__ void gmn_scan(const float* __restrict__ dbc, const float* __restrict__ dt,
                         const float* __restrict__ xc, const float* __restrict__ xz,
                         const float* __restrict__ Alog, const float* __restrict__ Dp,
                         float* __restrict__ ygate, int numseq, int T, int dir)
{
    using namespace cfg;
    const int idx = blockIdx.x * blockDim.x + threadIdx.x;
    if (idx >= numseq * DI) return;
    const int d = idx & (DI - 1);
    const int s = idx / DI;
    float h[DS], Ad[DS];
#pragma unroll
    for (int i = 0; i < DS; ++i) { h[i] = 0.f; Ad[i] = -__expf(Alog[d * DS + i]); }
    const float dp = Dp[d];
    for (int step = 0; step < T; ++step) {
        const int    t   = dir ? (T - 1 - step) : step;
        const size_t row = (size_t)s * T + t;
        const float dtv = dt[row * DI + d];
        const float xcv = xc[row * DI + d];
        const float zv  = xz[row * (2 * DI) + DI + d];
        const float* Bm = dbc + row * (R + 2 * DS) + R;
        const float* Cm = Bm + DS;
        float y = 0.f;
#pragma unroll
        for (int i = 0; i < DS; ++i) {
            h[i] = __expf(dtv * Ad[i]) * h[i] + dtv * Bm[i] * xcv;
            y += h[i] * Cm[i];
        }
        y += dp * xcv;
        const float sig = 1.f / (1.f + __expf(-zv));
        ygate[row * DI + d] = y * (zv * sig);
    }
}

// node[m,c] = hsum[(m*L + L-1)*D + c]
__global__ void gmn_take_last(const float* __restrict__ hsum, float* __restrict__ node)
{
    using namespace cfg;
    const int idx = blockIdx.x * blockDim.x + threadIdx.x;
    if (idx >= M1 * D) return;
    const int c = idx & (D - 1);
    const int m = idx / D;
    node[idx] = hsum[((size_t)m * L + (L - 1)) * D + c];
}

__global__ void gmn_degrees(const float* __restrict__ adj, float* __restrict__ deg)
{
    using namespace cfg;
    const int idx = blockIdx.x * blockDim.x + threadIdx.x;
    if (idx >= M1) return;
    const float* p = adj + (size_t)idx * N;
    float s = 0.f;
    for (int m = 0; m < N; ++m) s += p[m];
    deg[idx] = s;
}

// Stable ascending argsort by O(N^2) rank counting (matches jnp.argsort).
// perm[b, rank] = n ; inv[b, n] = rank.
__global__ void gmn_rank(const float* __restrict__ deg, int* __restrict__ perm,
                         int* __restrict__ inv)
{
    using namespace cfg;
    const int idx = blockIdx.x * blockDim.x + threadIdx.x;
    if (idx >= M1) return;
    const int b = idx / N, n = idx % N;
    const float* k = deg + (size_t)b * N;
    const float kn = k[n];
    int r = 0;
    for (int m = 0; m < N; ++m) {
        const float km = k[m];
        r += (km < kn) || (km == kn && m < n);
    }
    perm[b * N + r] = n;
    inv[idx] = r;
}

__global__ void gmn_unpermute(const float* __restrict__ hg, const int* __restrict__ inv,
                              float* __restrict__ ng)
{
    using namespace cfg;
    const int idx = blockIdx.x * blockDim.x + threadIdx.x;
    if (idx >= M1 * D) return;
    const int c   = idx & (D - 1);
    const int row = idx / D;
    const int b   = row / N;
    ng[idx] = hg[((size_t)b * N + inv[row]) * D + c];
}

// Bidirectional edge aggregation via f32 global atomics.
__global__ void gmn_edge_agg(const float* __restrict__ ng, const int* __restrict__ ei,
                             float* __restrict__ nei)
{
    using namespace cfg;
    const int idx = blockIdx.x * blockDim.x + threadIdx.x;
    if (idx >= B * 2 * E) return;
    const int b = idx / (2 * E);
    const int e = idx % (2 * E);
    int sa, da;
    if (e < E) { sa = ei[e];           da = ei[E + e]; }
    else       { sa = ei[E + (e - E)]; da = ei[e - E]; }
    const float* sp = ng + ((size_t)b * N + sa) * D;
    float*       dp = nei + ((size_t)b * N + da) * D;
#pragma unroll 8
    for (int c = 0; c < D; ++c) atomicAdd(&dp[c], sp[c]);
}

__global__ void gmn_add_relu(const float* __restrict__ ng, const float* __restrict__ mp,
                             float* __restrict__ ngf)
{
    using namespace cfg;
    const int idx = blockIdx.x * blockDim.x + threadIdx.x;
    if (idx >= M1 * D) return;
    ngf[idx] = ng[idx] + fmaxf(mp[idx], 0.f);
}

// ---------------------------------------------------------------------------
extern "C" void kernel_launch(void* const* d_in, const int* in_sizes, int n_in,
                              void* d_out, int out_size, void* d_ws, size_t ws_size,
                              hipStream_t stream)
{
    using namespace cfg;
    (void)in_sizes; (void)n_in; (void)out_size; (void)ws_size;

    const float* x         = (const float*)d_in[0];
    const float* adj       = (const float*)d_in[1];
    const int*   token_ids = (const int*)  d_in[2];
    const int*   edge_idx  = (const int*)  d_in[3];
    const float* enc_w1    = (const float*)d_in[4];
    const float* enc_b1    = (const float*)d_in[5];
    const float* enc_w2    = (const float*)d_in[6];
    const float* enc_b2    = (const float*)d_in[7];
    const float* ln_g      = (const float*)d_in[8];
    const float* ln_b      = (const float*)d_in[9];
    const float* in_proj_w = (const float*)d_in[10];
    const float* conv_w    = (const float*)d_in[11];
    const float* conv_b    = (const float*)d_in[12];
    const float* x_proj_w  = (const float*)d_in[13];
    const float* dt_proj_w = (const float*)d_in[14];
    const float* dt_proj_b = (const float*)d_in[15];
    const float* A_log     = (const float*)d_in[16];
    const float* D_param   = (const float*)d_in[17];
    const float* out_w     = (const float*)d_in[18];
    const float* mpnn_sw   = (const float*)d_in[19];
    const float* mpnn_sb   = (const float*)d_in[20];
    const float* mpnn_nw   = (const float*)d_in[21];
    const float* mpnn_nb   = (const float*)d_in[22];
    const float* proj_w    = (const float*)d_in[23];
    const float* proj_b    = (const float*)d_in[24];
    float*       out       = (float*)d_out;

    // ---- workspace layout (f32 elements) ----
    float* wsf = (float*)d_ws;
    size_t off = 0;
    auto take = [&](size_t n) { size_t o = off; off += (n + 63) & ~(size_t)63; return o; };
    const size_t oRegA = take((size_t)M0 * DI);          // pooled|tok1, later dt
    const size_t oXn   = take((size_t)M0 * D);
    const size_t oXz   = take((size_t)M0 * 2 * DI);
    const size_t oXc   = take((size_t)M0 * DI);
    const size_t oDbc  = take((size_t)M0 * (R + 2 * DS));
    const size_t oYg   = take((size_t)M0 * DI);
    const size_t oHs   = take((size_t)M0 * D);
    const size_t oNode = take((size_t)M1 * D);
    const size_t oXn1  = take((size_t)M1 * D);
    const size_t oHg   = take((size_t)M1 * D);
    const size_t oNg   = take((size_t)M1 * D);
    const size_t oNei  = take((size_t)M1 * D);
    const size_t oMp   = take((size_t)M1 * D);
    const size_t oNgf  = take((size_t)M1 * D);
    const size_t oDeg  = take((size_t)M1);
    const size_t oInt  = take((size_t)2 * M1);

    float* pooled = wsf + oRegA;
    float* tok1   = wsf + oRegA + (size_t)M0 * D;
    float* dtBuf  = wsf + oRegA;          // reuses pooled|tok1 after encoder
    float* xn     = wsf + oXn;
    float* xz     = wsf + oXz;
    float* xc     = wsf + oXc;
    float* dbc    = wsf + oDbc;
    float* yg     = wsf + oYg;
    float* hsum   = wsf + oHs;
    float* node   = wsf + oNode;
    float* xn1    = wsf + oXn1;
    float* hg     = wsf + oHg;
    float* ng     = wsf + oNg;
    float* nei    = wsf + oNei;
    float* mp     = wsf + oMp;
    float* ngf    = wsf + oNgf;
    float* deg    = wsf + oDeg;
    int*   perm   = (int*)(wsf + oInt);
    int*   inv    = perm + M1;

    auto cdiv = [](long a, long b) { return (int)((a + b - 1) / b); };

    // M must be a multiple of 64 (all call sites: 32768 or 4096); K in {64,128}.
    auto gemm = [&](const float* A, const float* W, const float* bias, float* C,
                    int M, int Nc, int K, float alpha, int act, int accf) {
        dim3 grid((Nc + 15) / 16, M / 64);
        if (act == 1)
            gmn_gemm<1, 0><<<grid, 128, 0, stream>>>(A, W, bias, C, M, Nc, K, alpha);
        else if (accf == 1)
            gmn_gemm<0, 1><<<grid, 128, 0, stream>>>(A, W, bias, C, M, Nc, K, alpha);
        else
            gmn_gemm<0, 0><<<grid, 128, 0, stream>>>(A, W, bias, C, M, Nc, K, alpha);
    };

    // One Mamba direction: in_proj -> conv+silu -> x_proj -> dt -> scan -> out_proj(acc 0.5)
    auto mamba_dir = [&](const float* xnBuf, int Mtok, int numseq, int T,
                         int j, int dir, float* hacc) {
        const float* Wi  = in_proj_w + (size_t)j * 2 * DI * D;
        const float* cw  = conv_w    + (size_t)j * DI * DC;
        const float* cb  = conv_b    + (size_t)j * DI;
        const float* Wx  = x_proj_w  + (size_t)j * (R + 2 * DS) * DI;
        const float* Wdt = dt_proj_w + (size_t)j * DI * R;
        const float* bdt = dt_proj_b + (size_t)j * DI;
        const float* Al  = A_log     + (size_t)j * DI * DS;
        const float* Dp  = D_param   + (size_t)j * DI;
        const float* Wo  = out_w     + (size_t)j * D * DI;
        gemm(xnBuf, Wi, nullptr, xz, Mtok, 2 * DI, D, 1.f, 0, 0);
        gmn_conv_silu<<<cdiv((long)Mtok * DI, 256), 256, 0, stream>>>(xz, cw, cb, xc, Mtok, T, dir);
        gemm(xc, Wx, nullptr, dbc, Mtok, R + 2 * DS, DI, 1.f, 0, 0);
        gmn_dt_softplus<<<cdiv((long)Mtok * DI, 256), 256, 0, stream>>>(dbc, Wdt, bdt, dtBuf, Mtok);
        gmn_scan<<<cdiv((long)numseq * DI, 256), 256, 0, stream>>>(dbc, dtBuf, xc, xz, Al, Dp,
                                                                   yg, numseq, T, dir);
        gemm(yg, Wo, nullptr, hacc, Mtok, D, DI, 0.5f, 0, 1);
    };

    // ---- Stage 1: pooling + token encoder ----
    gmn_pool<<<cdiv((long)M0 * H, 256), 256, 0, stream>>>(x, token_ids, pooled);
    gemm(pooled, enc_w1, enc_b1, tok1, M0, D, H, 1.f, /*relu*/1, 0);
    gemm(tok1, enc_w2, enc_b2, xn, M0, D, D, 1.f, 0, 0);

    // ---- Stage 2: sequence-level bidirectional Mamba (LN in place) ----
    gmn_layernorm<<<cdiv(M0, 256), 256, 0, stream>>>(xn, xn, ln_g, ln_b, M0);
    (void)hipMemsetAsync(hsum, 0, (size_t)M0 * D * sizeof(float), stream);
    mamba_dir(xn, M0, B * N, L, 0, 0, hsum);
    mamba_dir(xn, M0, B * N, L, 1, 1, hsum);
    gmn_take_last<<<cdiv((long)M1 * D, 256), 256, 0, stream>>>(hsum, node);

    // ---- Stage 3: degree sort + graph-level bidirectional Mamba ----
    gmn_degrees<<<cdiv(M1, 256), 256, 0, stream>>>(adj, deg);
    gmn_rank<<<cdiv(M1, 256), 256, 0, stream>>>(deg, perm, inv);
    gmn_gather_ln<<<cdiv(M1, 256), 256, 0, stream>>>(node, perm, ln_g + D, ln_b + D, xn1);
    (void)hipMemsetAsync(hg, 0, (size_t)M1 * D * sizeof(float), stream);
    mamba_dir(xn1, M1, B, N, 2, 0, hg);
    mamba_dir(xn1, M1, B, N, 3, 1, hg);
    gmn_unpermute<<<cdiv((long)M1 * D, 256), 256, 0, stream>>>(hg, inv, ng);

    // ---- Stage 4: MPNN + output projection ----
    (void)hipMemsetAsync(nei, 0, (size_t)M1 * D * sizeof(float), stream);
    gmn_edge_agg<<<cdiv((long)B * 2 * E, 256), 256, 0, stream>>>(ng, edge_idx, nei);
    gemm(ng,  mpnn_sw, mpnn_sb, mp, M1, D, D, 1.f, 0, 0);
    gemm(nei, mpnn_nw, mpnn_nb, mp, M1, D, D, 1.f, 0, 1);
    gmn_add_relu<<<cdiv((long)M1 * D, 256), 256, 0, stream>>>(ng, mp, ngf);
    gemm(ngf, proj_w, proj_b, out, M1, 2 * D, D, 1.f, 0, 0);
}
